// GPT2Attention_15857019257268
// MI455X (gfx1250) — compile-verified, matrix operand order checked by reading, not run on hardware
//
#include <hip/hip_runtime.h>
#include <hip/hip_bf16.h>

// GPT-2 attention, MI455X (gfx1250): f16 WMMA pipeline, TDM (tensor DMA)
// double-buffered tile feed, permlane16 softmax reductions.
// B=8, T=1024, D=768, NH=12, HD=64.

#define Bsz 8
#define Tt  1024
#define Dd  768
#define NHh 12
#define HDd 64
#define C3  2304   // 3*D

typedef __attribute__((ext_vector_type(16))) _Float16     v16h;
typedef __attribute__((ext_vector_type(8)))  float        v8f;
typedef __attribute__((ext_vector_type(4)))  unsigned int v4u;
typedef __attribute__((ext_vector_type(8)))  int          v8i;
typedef __attribute__((ext_vector_type(4)))  int          v4i;

#ifndef __has_builtin
#define __has_builtin(x) 0
#endif

#if __has_builtin(__builtin_amdgcn_tensor_load_to_lds) && \
    __has_builtin(__builtin_amdgcn_s_wait_tensorcnt)
#define HAS_TDM 1
#else
#define HAS_TDM 0
#endif

#if __has_builtin(__builtin_amdgcn_permlane16)
#define HAS_PERMLANE 1
#else
#define HAS_PERMLANE 0
#endif

__device__ __forceinline__ unsigned short f2h(float f) {
    union { _Float16 h; unsigned short u; } c;
    c.h = (_Float16)f;
    return c.u;
}

// 16-bit WMMA fragment gather (A- or B-layout): lane = row (M or N), per-lane
// regs hold K per the CDNA5 ISA table.  Element (row,k) at p[row*stride + k].
__device__ __forceinline__ v16h frag_ld(const unsigned short* p, int row,
                                        int stride, int lhalf) {
    union { v16h v; unsigned int d[8]; } u;
    const unsigned short* r = p + row * stride;
#pragma unroll
    for (int j = 0; j < 8; ++j) {
        int kp = 2 * (j & 3) + 8 * lhalf + 16 * (j >> 2);
        u.d[j] = *(const unsigned int*)(r + kp);
    }
    return u.v;
}

__device__ __forceinline__ v8f wmma16(v16h a, v16h b, v8f c) {
    return __builtin_amdgcn_wmma_f32_16x16x32_f16(false, a, false, b,
                                                  (short)0, c, false, false);
}

// ---- cross-lane reductions within 16-lane groups (rows of the C layout) ----
#if HAS_PERMLANE
__device__ __forceinline__ float xor16(float x, int s0, int s1) {
    int v = __float_as_int(x);
    return __int_as_float(__builtin_amdgcn_permlane16(v, v, s0, s1, false, false));
}
__device__ __forceinline__ float rmax16(float x) {
    x = fmaxf(x, xor16(x, 0x67452301, 0xEFCDAB89));   // ^1
    x = fmaxf(x, xor16(x, 0x54761032, 0xDCFE98BA));   // ^2
    x = fmaxf(x, xor16(x, 0x32107654, 0xBA98FEDC));   // ^4
    x = fmaxf(x, xor16(x, 0xFEDCBA98, 0x76543210));   // ^8
    return x;
}
__device__ __forceinline__ float rsum16(float x) {
    x += xor16(x, 0x67452301, 0xEFCDAB89);
    x += xor16(x, 0x54761032, 0xDCFE98BA);
    x += xor16(x, 0x32107654, 0xBA98FEDC);
    x += xor16(x, 0xFEDCBA98, 0x76543210);
    return x;
}
#else
__device__ __forceinline__ float rmax16(float x) {
    for (int mk = 1; mk < 16; mk <<= 1) x = fmaxf(x, __shfl_xor(x, mk, 32));
    return x;
}
__device__ __forceinline__ float rsum16(float x) {
    for (int mk = 1; mk < 16; mk <<= 1) x += __shfl_xor(x, mk, 32);
    return x;
}
#endif

// ---- Tensor Data Mover: 2D tile load, LDS padding folded into descriptor ----
// t0 halfwords per row, t1 rows, row pitch stride0 halfwords.  pic/pac chosen
// so LDS row stride = 2^(pic+1) + (pac+1) dwords.
#if HAS_TDM
__device__ __forceinline__ void tdm_load_2d(const unsigned short* g,
                                            unsigned lds_off,
                                            unsigned t0, unsigned t1,
                                            unsigned long long stride0,
                                            unsigned pic, unsigned pac) {
    unsigned long long ga = (unsigned long long)(uintptr_t)g;
    v4u g0;
    g0[0] = 1u;                                          // count=1, user D#
    g0[1] = lds_off;                                     // LDS byte address
    g0[2] = (unsigned)(ga & 0xFFFFFFFFu);                // global addr lo
    g0[3] = (unsigned)((ga >> 32) & 0x01FFFFFFu) | 0x80000000u;  // hi | type=2
    v8i g1;
    g1[0] = (int)((1u << 16) |               // data_size = 2 bytes
                  (1u << 20) |               // pad_enable
                  (pic << 22) | (pac << 25));
    g1[1] = (int)((t0 & 0xFFFFu) << 16);                 // tensor_dim0 lo
    g1[2] = (int)((t0 >> 16) | ((t1 & 0xFFFFu) << 16));  // dim0 hi | dim1 lo
    g1[3] = (int)((t1 >> 16) | (t0 << 16));              // dim1 hi | tile_dim0
    g1[4] = (int)(t1 & 0xFFFFu);                         // tile_dim1, tile_dim2=0
    g1[5] = (int)(unsigned)(stride0 & 0xFFFFFFFFu);      // dim0 stride lo
    g1[6] = (int)(unsigned)((stride0 >> 32) & 0xFFFFu);  // dim0 stride hi
    g1[7] = 0;
    v4i gz = {0, 0, 0, 0};
#if defined(__clang_major__) && __clang_major__ >= 23
    v8i gz8 = {0, 0, 0, 0, 0, 0, 0, 0};
    __builtin_amdgcn_tensor_load_to_lds(g0, g1, gz, gz, gz8, 0);
#else
    __builtin_amdgcn_tensor_load_to_lds(g0, g1, gz, gz, 0);
#endif
}
#endif

// ---------------------------------------------------------------- convert ---
__global__ __launch_bounds__(256) void cvtk(const float* __restrict__ s,
                                            unsigned short* __restrict__ d,
                                            int n) {
    int i = (blockIdx.x * 256 + threadIdx.x) * 4;
    if (i < n) {
        float4 f = *(const float4*)(s + i);
        unsigned int p0 = (unsigned int)f2h(f.x) | ((unsigned int)f2h(f.y) << 16);
        unsigned int p1 = (unsigned int)f2h(f.z) | ((unsigned int)f2h(f.w) << 16);
        *(unsigned int*)(d + i)     = p0;
        *(unsigned int*)(d + i + 2) = p1;
    }
}

// ------------------------------------------------------------------- GEMM ---
// C[M,N] = A[M,K] * B[N,K]^T + bias.  Block 128x256, BK=32, 8 waves (2x4),
// wave tile 64x64 = 16 WMMA accumulators.  Tiles DMA'd by the TDM into LDS
// (pad descriptor yields stride-40 rows: 16 distinct banks for frag gathers),
// double-buffered, synchronized with s_wait_tensorcnt + barriers.
#define TA (128 * 40)
#define TB (256 * 40)

template <bool OUTF32>
__global__ __launch_bounds__(256) void gemm_h(
    const unsigned short* __restrict__ A,
    const unsigned short* __restrict__ Bm,
    const float* __restrict__ bias,
    float* __restrict__ Cf,
    unsigned short* __restrict__ Ch,
    int M, int N, int K) {
    __shared__ unsigned short Sm[2 * (TA + TB)];
    const int tid = threadIdx.x, wave = tid >> 5, lane = tid & 31;
    const int l16 = lane & 15, lhalf = lane >> 4;
    const int m0 = blockIdx.y * 128, n0 = blockIdx.x * 256;
    const int wm = (wave >> 2) * 64, wn = (wave & 3) * 64;
    const int NK = K / 32;

    v8f acc[4][4];
#pragma unroll
    for (int i = 0; i < 4; ++i)
#pragma unroll
        for (int j = 0; j < 4; ++j)
#pragma unroll
            for (int v = 0; v < 8; ++v) acc[i][j][v] = 0.f;

#if HAS_TDM
    if (wave == 0) {   // prologue: DMA first tiles into buffer 0
        tdm_load_2d(A  + (size_t)m0 * K, (unsigned)(uintptr_t)Sm,
                    32, 128, (unsigned long long)K, 3, 3);
        tdm_load_2d(Bm + (size_t)n0 * K, (unsigned)(uintptr_t)(Sm + TA),
                    32, 256, (unsigned long long)K, 3, 3);
    }
#endif
    for (int ki = 0; ki < NK; ++ki) {
        const int cur = ki & 1;
        unsigned short* As = Sm + cur * (TA + TB);
        unsigned short* Bs = As + TA;
#if HAS_TDM
        __syncthreads();   // buf[nxt] readers from iter ki-1 are done
        if (wave == 0) {
            if (ki + 1 < NK) {
                unsigned short* An = Sm + (cur ^ 1) * (TA + TB);
                tdm_load_2d(A  + (size_t)m0 * K + (ki + 1) * 32,
                            (unsigned)(uintptr_t)An, 32, 128,
                            (unsigned long long)K, 3, 3);
                tdm_load_2d(Bm + (size_t)n0 * K + (ki + 1) * 32,
                            (unsigned)(uintptr_t)(An + TA), 32, 256,
                            (unsigned long long)K, 3, 3);
                __builtin_amdgcn_s_wait_tensorcnt(2);  // current pair landed
            } else {
                __builtin_amdgcn_s_wait_tensorcnt(0);
            }
        }
        __syncthreads();   // completion visible to all waves
#else
        __syncthreads();
        for (int idx = tid; idx < 128 * 16; idx += 256) {   // A tile (dwords)
            int r = idx >> 4, c = (idx & 15) * 2;
            *(unsigned int*)(As + r * 40 + c) =
                *(const unsigned int*)(A + (size_t)(m0 + r) * K + ki * 32 + c);
        }
        for (int idx = tid; idx < 256 * 16; idx += 256) {   // B tile
            int r = idx >> 4, c = (idx & 15) * 2;
            *(unsigned int*)(Bs + r * 40 + c) =
                *(const unsigned int*)(Bm + (size_t)(n0 + r) * K + ki * 32 + c);
        }
        __syncthreads();
#endif
        v16h bf[4];
#pragma unroll
        for (int ns = 0; ns < 4; ++ns)
            bf[ns] = frag_ld(Bs, wn + ns * 16 + l16, 40, lhalf);
#pragma unroll
        for (int ms = 0; ms < 4; ++ms) {
            v16h af = frag_ld(As, wm + ms * 16 + l16, 40, lhalf);
#pragma unroll
            for (int ns = 0; ns < 4; ++ns)
                acc[ms][ns] = wmma16(af, bf[ns], acc[ms][ns]);
        }
    }

#pragma unroll
    for (int ms = 0; ms < 4; ++ms)
#pragma unroll
        for (int ns = 0; ns < 4; ++ns) {
            int gn = n0 + wn + ns * 16 + l16;
            float bv = bias[gn];
#pragma unroll
            for (int v = 0; v < 8; ++v) {
                int gm = m0 + wm + ms * 16 + v + 8 * lhalf;
                float val = acc[ms][ns][v] + bv;
                if (OUTF32) Cf[(size_t)gm * N + gn] = val;
                else        Ch[(size_t)gm * N + gn] = f2h(val);
            }
        }
}

// -------------------------------------------------------------- attention ---
// Block = (q-block of 128 rows, b*h).  Wave = 16 q rows.  K tiles DMA'd by
// TDM (double-buffered, pad descriptor -> stride-72 rows), V staged transposed
// by the threads, flash softmax with permlane16 row reductions.
#define KSZ (32 * 72)

__global__ __launch_bounds__(256) void attn_h(
    const unsigned short* __restrict__ qkv,   // [B*T, 3D] f16
    unsigned short* __restrict__ yb) {        // [B*T, D]  f16
    __shared__ unsigned short Ks[2 * KSZ];        // [key][hd] stride 72
    __shared__ unsigned short Vs[64 * 34];        // [hd][key] (transposed)
    __shared__ unsigned short Ps[8 * 16 * 34];    // per-wave P tile
    const int tid = threadIdx.x, wave = tid >> 5, lane = tid & 31;
    const int l16 = lane & 15, lhalf = lane >> 4;
    const int qb = blockIdx.x;
    const int b = blockIdx.y / NHh, h = blockIdx.y % NHh;
    const int tq0  = qb * 128 + wave * 16;
    const int grow = b * Tt + tq0;

    // Q fragments, pre-scaled by HD^-1/2 = 0.125 (exact in f16)
    const unsigned short* qbase = qkv + h * HDd;
    v16h q0 = frag_ld(qbase,      grow + l16, C3, lhalf);
    v16h q1 = frag_ld(qbase + 32, grow + l16, C3, lhalf);
#pragma unroll
    for (int e = 0; e < 16; ++e) {
        q0[e] *= (_Float16)0.125f;
        q1[e] *= (_Float16)0.125f;
    }

    float mi[8], li[8];
#pragma unroll
    for (int v = 0; v < 8; ++v) { mi[v] = -1e30f; li[v] = 0.f; }
    v8f acc[4];
#pragma unroll
    for (int n4 = 0; n4 < 4; ++n4)
#pragma unroll
        for (int v = 0; v < 8; ++v) acc[n4][v] = 0.f;

    const int nblk = 4 * qb + 4;
    const int trow = tid >> 3, tcol = (tid & 7) * 8;

#if HAS_TDM
    if (wave == 0)   // prologue: DMA first K tile
        tdm_load_2d(qkv + (size_t)(b * Tt) * C3 + Dd + h * HDd,
                    (unsigned)(uintptr_t)Ks, 64, 32, (unsigned long long)C3, 4, 3);
#endif

    for (int j = 0; j < nblk; ++j) {
        const int tk0 = j * 32;
        unsigned short* Kcur = Ks + (j & 1) * KSZ;
        // V rows (global, coalesced) fetched before the barrier for overlap
        uint4 vv = *(const uint4*)(qkv + (size_t)(b * Tt + tk0 + trow) * C3 +
                                   2 * Dd + h * HDd + tcol);
        __syncthreads();   // prior iter done reading Vs and Ks[nxt]
#if HAS_TDM
        if (wave == 0) {
            if (j + 1 < nblk)
                tdm_load_2d(qkv + (size_t)(b * Tt + tk0 + 32) * C3 + Dd + h * HDd,
                            (unsigned)(uintptr_t)(Ks + ((j & 1) ^ 1) * KSZ),
                            64, 32, (unsigned long long)C3, 4, 3);
        }
#else
        for (int idx = tid; idx < 32 * 32; idx += 256) {   // K tile (dwords)
            int r = idx >> 5, c = (idx & 31) * 2;
            *(unsigned int*)(Kcur + r * 72 + c) =
                *(const unsigned int*)(qkv + (size_t)(b * Tt + tk0 + r) * C3 +
                                       Dd + h * HDd + c);
        }
#endif
        {   // stage V transposed
            union { uint4 q; unsigned short u[8]; } vu; vu.q = vv;
#pragma unroll
            for (int e = 0; e < 8; ++e) Vs[(tcol + e) * 34 + trow] = vu.u[e];
        }
#if HAS_TDM
        if (wave == 0) {
            if (j + 1 < nblk) __builtin_amdgcn_s_wait_tensorcnt(1);
            else              __builtin_amdgcn_s_wait_tensorcnt(0);
        }
#endif
        __syncthreads();

        if (tk0 <= tq0 + 15) {                 // wave-uniform causal skip
            v8f s0, s1;
#pragma unroll
            for (int v = 0; v < 8; ++v) { s0[v] = 0.f; s1[v] = 0.f; }
            v16h k00 = frag_ld(Kcur,      l16,      72, lhalf);
            v16h k01 = frag_ld(Kcur + 32, l16,      72, lhalf);
            v16h k10 = frag_ld(Kcur,      16 + l16, 72, lhalf);
            v16h k11 = frag_ld(Kcur + 32, 16 + l16, 72, lhalf);
            s0 = wmma16(q0, k00, s0); s0 = wmma16(q1, k01, s0);
            s1 = wmma16(q0, k10, s1); s1 = wmma16(q1, k11, s1);

            if (tk0 + 31 > tq0) {              // only the diagonal block masks
#pragma unroll
                for (int v = 0; v < 8; ++v) {
                    int tq = tq0 + v + 8 * lhalf;
                    if (tk0 + l16      > tq) s0[v] = -1e30f;
                    if (tk0 + 16 + l16 > tq) s1[v] = -1e30f;
                }
            }
            float al[8];
#pragma unroll
            for (int v = 0; v < 8; ++v) {
                float mx = rmax16(fmaxf(s0[v], s1[v]));
                float mn = fmaxf(mi[v], mx);
                al[v] = __expf(mi[v] - mn);
                mi[v] = mn;
                float p0 = __expf(s0[v] - mn), p1 = __expf(s1[v] - mn);
                s0[v] = p0; s1[v] = p1;
                li[v] = li[v] * al[v] + rsum16(p0 + p1);
            }
#pragma unroll
            for (int n4 = 0; n4 < 4; ++n4)
#pragma unroll
                for (int v = 0; v < 8; ++v) acc[n4][v] *= al[v];

            // restage P: C layout -> LDS -> A layout fragment
            unsigned short* Pw = Ps + wave * (16 * 34);
#pragma unroll
            for (int v = 0; v < 8; ++v) {
                int pm = v + 8 * lhalf;
                Pw[pm * 34 + l16]      = f2h(s0[v]);
                Pw[pm * 34 + 16 + l16] = f2h(s1[v]);
            }
            asm volatile("" ::: "memory");     // keep DS store->load order
            v16h pf = frag_ld(Pw, l16, 34, lhalf);
#pragma unroll
            for (int n4 = 0; n4 < 4; ++n4) {
                v16h vf = frag_ld(Vs, n4 * 16 + l16, 34, lhalf);
                acc[n4] = wmma16(pf, vf, acc[n4]);
            }
        }
    }

#pragma unroll
    for (int n4 = 0; n4 < 4; ++n4)
#pragma unroll
        for (int v = 0; v < 8; ++v) {
            int m = v + 8 * lhalf;
            float y = acc[n4][v] / li[v];
            yb[(size_t)(grow + m) * Dd + h * HDd + n4 * 16 + l16] = f2h(y);
        }
}

// ----------------------------------------------------------------- launch ---
extern "C" void kernel_launch(void* const* d_in, const int* in_sizes, int n_in,
                              void* d_out, int out_size, void* d_ws, size_t ws_size,
                              hipStream_t stream) {
    const float* x      = (const float*)d_in[0];
    const float* w_attn = (const float*)d_in[1];
    const float* b_attn = (const float*)d_in[2];
    const float* w_proj = (const float*)d_in[3];
    const float* b_proj = (const float*)d_in[4];
    float* out = (float*)d_out;

    unsigned short* ws = (unsigned short*)d_ws;
    size_t o = 0;
    unsigned short* xb   = ws + o; o += (size_t)8192 * 768;    // x    f16
    unsigned short* wab  = ws + o; o += (size_t)2304 * 768;    // Wqkv f16
    unsigned short* wpb  = ws + o; o += (size_t)768 * 768;     // Wout f16
    unsigned short* qkvb = ws + o; o += (size_t)8192 * 2304;   // qkv  f16
    unsigned short* yb   = ws + o; o += (size_t)8192 * 768;    // attn f16

    cvtk<<<dim3((8192 * 768) / 1024), 256, 0, stream>>>(x, xb, 8192 * 768);
    cvtk<<<dim3((2304 * 768) / 1024), 256, 0, stream>>>(w_attn, wab, 2304 * 768);
    cvtk<<<dim3((768 * 768) / 1024),  256, 0, stream>>>(w_proj, wpb, 768 * 768);

    // qkv = x @ Wqkv^T + b  (f16 out)
    gemm_h<false><<<dim3(2304 / 256, 8192 / 128), 256, 0, stream>>>(
        xb, wab, b_attn, nullptr, qkvb, 8192, 2304, 768);

    // flash attention per (q-block, b*h)
    attn_h<<<dim3(8, Bsz * NHh), 256, 0, stream>>>(qkvb, yb);

    // out = y @ Wout^T + b  (f32 out)
    gemm_h<true><<<dim3(768 / 256, 8192 / 128), 256, 0, stream>>>(
        yb, wpb, b_proj, out, nullptr, 8192, 768, 768);
}
